// ViT_81174881894813
// MI455X (gfx1250) — matrix operand order repaired and checked by reading, not compiled, not run on hardware
//
#include <hip/hip_runtime.h>
#include <math.h>

typedef __attribute__((ext_vector_type(16))) __bf16 v16bf;
typedef __attribute__((ext_vector_type(2)))  __bf16 bf16x2;
typedef __attribute__((ext_vector_type(8)))  float  v8f;

#define BATCH 16
#define TOK   197
#define DMODEL 768
#define NHEAD 12
#define HDIM  64
#define MLPD  3072
#define SLD   208   // padded score row stride (divisible by 16)

__device__ __forceinline__ __bf16 f2bf(float f) { return (__bf16)f; }  // native v_cvt, RNE

// Pack two floats into one dword of bf16 pair -> single v_cvt_pk_bf16_f32
__device__ __forceinline__ bf16x2 pk2(float lo, float hi) {
    bf16x2 r; r.x = (__bf16)lo; r.y = (__bf16)hi; return r;
}

union Frag16 {
    v16bf v;
    uint4 q[2];
    __bf16 s[16];
};

union Stage16 {
    bf16x2 h[8];
    uint4  q[2];
    __bf16 s[16];
};

// ---------------------------------------------------------------------------
// Tiled WMMA GEMM: C[M,N] = alpha * A[M,K] * B + bias + (gelu) + resid
//   A: fp32 row-major, lda % 4 == 0.
//   B: if B_NK: stored [N][K] (K contiguous), bf16 (B_BF16) or fp32
//      else   : stored [K][N] (N contiguous), fp32 (transposed while staging)
//   Batched via grid.z: z = zb*nh + zh, per-operand (b,h) strides.
// Block: 256 threads (8 wave32), tile 128x128, K-step 32, double-buffered LDS.
// Each wave computes a 32x64 sub-tile: 2x4 fragments -> 8 WMMA per K-step.
// OOB policy: row/col indices CLAMPED (garbage masked at store); only the
// K-tail is zero-filled. The full-tile/tail choice is BLOCK-UNIFORM
// (k0+32<=K) so the hot loop has no exec divergence; all per-thread staging
// base addresses are hoisted out of the K loop.
// ---------------------------------------------------------------------------
template<bool B_BF16, bool B_NK, bool DO_GELU>
__global__ __launch_bounds__(256) void gemm_wmma_bf16(
    const float* __restrict__ A, int lda, long sAb, long sAh,
    const void*  __restrict__ Bp, int ldb, long sBb, long sBh,
    const float* __restrict__ bias,
    const float* __restrict__ resid, int ldr, long sRb,
    float* __restrict__ C, int ldc, long sCb, long sCh,
    int M, int N, int K, float alpha, int nh)
{
    __shared__ __bf16 lsA[2][128][40];   // 128 rows x 32 k (80B rows, 16B-aligned chunks)
    __shared__ __bf16 lsB[2][128][40];   // 128 cols x 32 k

    const int z  = blockIdx.z;
    const int zb = z / nh, zh = z % nh;
    A += (long)zb * sAb + (long)zh * sAh;
    C += (long)zb * sCb + (long)zh * sCh;
    const long boff = (long)zb * sBb + (long)zh * sBh;

    const int tid  = threadIdx.x;
    const int wave = tid >> 5;
    const int lane = tid & 31;
    const int wr = (wave & 3) * 32;            // wave row offset within tile (0,32,64,96)
    const int wc = (wave >> 2) * 64;           // wave col offset within tile (0,64)
    const int m0 = blockIdx.y * 128;
    const int n0 = blockIdx.x * 128;

    const v8f vzero = {0.f,0.f,0.f,0.f,0.f,0.f,0.f,0.f};
    v8f acc[2][4] = {{vzero,vzero,vzero,vzero},{vzero,vzero,vzero,vzero}};

    // ---- loop-invariant staging bases (clamped; hoisted out of K loop) ----
    // A: thread covers row (tid>>1), k half (tid&1)*16
    const int sRow = tid >> 1;
    const int sKb  = (tid & 1) * 16;
    const int gmA  = m0 + sRow;
    const long aRow = (long)(gmA < M ? gmA : M - 1) * lda;  // clamped row offset
    const float* aBase = A + aRow + sKb;                    // + k0 at use (full path)

    // B (NK layout): thread covers col (tid>>1), k half (tid&1)*16
    const int gnB  = n0 + sRow;
    const long bRow = (long)(gnB < N ? gnB : N - 1) * ldb;
    const __bf16* bBaseH = B_BF16 ? ((const __bf16*)Bp + boff + bRow + sKb) : nullptr;
    const float*  bBaseF = !B_BF16 ? ((const float*)Bp + boff + bRow + sKb) : nullptr;

    // B (KN layout): thread covers k row (tid>>3), 16 cols at (tid&7)*16
    const int kkB = tid >> 3;
    const int nbB = (tid & 7) * 16;
    int ncOff[16];            // clamped column offsets (loop-invariant)
    bool nFull = (n0 + nbB + 15) < N;
    if (!B_NK) {
        #pragma unroll
        for (int c = 0; c < 16; ++c) {
            int gn = n0 + nbB + c;
            ncOff[c] = gn < N ? gn : N - 1;
        }
    }
    const float* bKNBase = (const float*)Bp + boff;

    auto stageA = [&](int buf, int k0, bool full) {
        Stage16 tmp;
        if (full) {
            const float* p = aBase + k0;
            #pragma unroll
            for (int c = 0; c < 4; ++c) {
                float4 f = *(const float4*)(p + c * 4);
                tmp.h[c*2+0] = pk2(f.x, f.y);
                tmp.h[c*2+1] = pk2(f.z, f.w);
            }
        } else {
            const float* rp = A + aRow;
            #pragma unroll
            for (int c = 0; c < 16; ++c) {
                int k  = k0 + sKb + c;
                int kc = k < K ? k : K - 1;
                float f = rp[kc];
                tmp.s[c] = f2bf(k < K ? f : 0.0f);   // K tail must be zero
            }
        }
        uint4* dst = (uint4*)&lsA[buf][sRow][sKb];
        dst[0] = tmp.q[0];
        dst[1] = tmp.q[1];
    };

    auto stageB = [&](int buf, int k0, bool full) {
        if (B_NK) {
            Stage16 tmp;
            if (B_BF16) {
                if (full) {
                    const uint4* bp4 = (const uint4*)(bBaseH + k0);
                    tmp.q[0] = bp4[0];
                    tmp.q[1] = bp4[1];
                } else {
                    const __bf16* rp = (const __bf16*)Bp + boff + bRow;
                    #pragma unroll
                    for (int c = 0; c < 16; ++c) {
                        int k  = k0 + sKb + c;
                        int kc = k < K ? k : K - 1;
                        __bf16 hv = rp[kc];
                        tmp.s[c] = k < K ? hv : (__bf16)0.0f;
                    }
                }
            } else {
                if (full) {
                    const float* p = bBaseF + k0;
                    #pragma unroll
                    for (int c = 0; c < 4; ++c) {
                        float4 f = *(const float4*)(p + c * 4);
                        tmp.h[c*2+0] = pk2(f.x, f.y);
                        tmp.h[c*2+1] = pk2(f.z, f.w);
                    }
                } else {
                    const float* rp = (const float*)Bp + boff + bRow;
                    #pragma unroll
                    for (int c = 0; c < 16; ++c) {
                        int k  = k0 + sKb + c;
                        int kc = k < K ? k : K - 1;
                        float f = rp[kc];
                        tmp.s[c] = f2bf(k < K ? f : 0.0f);
                    }
                }
            }
            uint4* dst = (uint4*)&lsB[buf][sRow][sKb];
            dst[0] = tmp.q[0];
            dst[1] = tmp.q[1];
        } else {
            // B stored [K][N] fp32 (V matrix): transpose while staging.
            const int gk  = k0 + kkB;
            const int gkc = gk < K ? gk : K - 1;
            const bool kok = gk < K;                 // K tail must be zero
            const float* rp = bKNBase + (long)gkc * ldb;
            if (nFull) {
                const float* p = rp + n0 + nbB;
                #pragma unroll
                for (int c = 0; c < 4; ++c) {
                    float4 f = *(const float4*)(p + c * 4);
                    lsB[buf][nbB + c*4+0][kkB] = f2bf(kok ? f.x : 0.0f);
                    lsB[buf][nbB + c*4+1][kkB] = f2bf(kok ? f.y : 0.0f);
                    lsB[buf][nbB + c*4+2][kkB] = f2bf(kok ? f.z : 0.0f);
                    lsB[buf][nbB + c*4+3][kkB] = f2bf(kok ? f.w : 0.0f);
                }
            } else {
                #pragma unroll
                for (int c = 0; c < 16; ++c) {
                    float f = rp[ncOff[c]];
                    lsB[buf][nbB + c][kkB] = f2bf(kok ? f : 0.0f);
                }
            }
        }
    };

    const int nk = (K + 31) >> 5;
    stageA(0, 0, 32 <= K);
    stageB(0, 0, 32 <= K);
    __syncthreads();

    const int rrow = lane & 15;
    const int asel = (lane < 16) ? 0 : 8;    // A: lanes0-15 K{0..7,16..23}, lanes16-31 K{8..15,24..31}
    const int bsel = (lane < 16) ? 0 : 8;    // B: lanes0-15 K{0..7,16..23}, lanes16-31 K{8..15,24..31}

    for (int t = 0; t < nk; ++t) {
        const int cur = t & 1;
        if (t + 1 < nk) {                     // prefetch next tile into the other buffer
            const int k1 = (t + 1) * 32;
            const bool full = (k1 + 32) <= K; // block-uniform branch
            stageA(cur ^ 1, k1, full);
            stageB(cur ^ 1, k1, full);
        }
        Frag16 af[2], bfm[4];
        #pragma unroll
        for (int i = 0; i < 2; ++i) {
            const __bf16* p = &lsA[cur][wr + 16 * i + rrow][0];
            af[i].q[0] = *(const uint4*)(p + asel);
            af[i].q[1] = *(const uint4*)(p + asel + 16);
        }
        #pragma unroll
        for (int j = 0; j < 4; ++j) {
            const __bf16* p = &lsB[cur][wc + 16 * j + rrow][0];
            bfm[j].q[0] = *(const uint4*)(p + bsel);
            bfm[j].q[1] = *(const uint4*)(p + bsel + 16);
        }
        #pragma unroll
        for (int i = 0; i < 2; ++i)
            #pragma unroll
            for (int j = 0; j < 4; ++j)
                acc[i][j] = __builtin_amdgcn_wmma_f32_16x16x32_bf16(
                    false, af[i].v, false, bfm[j].v, (short)0, acc[i][j], false, false);
        __syncthreads();
    }

    // ---- epilogue: alpha, bias, gelu, residual; per-fragment uniform bounds ----
    const int mlo = 8 * (lane >> 4);
    const int nlo = lane & 15;
    const float* rBase = resid ? (resid + (long)zb * sRb) : nullptr;
    #pragma unroll
    for (int i = 0; i < 2; ++i) {
        #pragma unroll
        for (int j = 0; j < 4; ++j) {
            const int mBase = m0 + wr + 16 * i + mlo;   // rows mBase..mBase+7
            const int nn    = n0 + wc + 16 * j + nlo;
            const bool fragFull = ((m0 + wr + 16 * i + 16) <= M) &&
                                  ((n0 + wc + 16 * j + 16) <= N);   // wave-uniform
            const float bv = (bias && nn < N) ? bias[nn] : 0.0f;
            if (fragFull) {
                float* cp = C + (long)mBase * ldc + nn;
                const float* rp = rBase ? (rBase + (long)mBase * ldr + nn) : nullptr;
                #pragma unroll
                for (int e = 0; e < 8; ++e) {
                    float v = acc[i][j][e] * alpha + bv;
                    if (DO_GELU) v = 0.5f * v * (1.0f + erff(v * 0.70710678118f));
                    if (resid) v += rp[(long)e * ldr];
                    cp[(long)e * ldc] = v;
                }
            } else {
                #pragma unroll
                for (int e = 0; e < 8; ++e) {
                    int m = mBase + e;
                    if (m < M && nn < N) {
                        float v = acc[i][j][e] * alpha + bv;
                        if (DO_GELU) v = 0.5f * v * (1.0f + erff(v * 0.70710678118f));
                        if (resid) v += rBase[(long)m * ldr + nn];
                        C[(long)m * ldc + nn] = v;
                    }
                }
            }
        }
    }
}

// ---------------------------------------------------------------------------
// Weight convert + transpose: W[K][N] fp32 -> Wt[N][K] bf16
// ---------------------------------------------------------------------------
__global__ __launch_bounds__(256) void convert_transpose_k(
    const float* __restrict__ W, __bf16* __restrict__ Wt, int K, int N)
{
    long idx = (long)blockIdx.x * 256 + threadIdx.x;
    if (idx >= (long)K * N) return;
    int n = (int)(idx % N);
    long k = idx / N;
    Wt[(long)n * K + k] = f2bf(W[idx]);
}

// ---------------------------------------------------------------------------
// LayerNorm: one block per row
// ---------------------------------------------------------------------------
__global__ __launch_bounds__(256) void layernorm_k(
    const float* __restrict__ X, long xStride,
    float* __restrict__ Y, long yStride,
    const float* __restrict__ g, const float* __restrict__ b, int n)
{
    __shared__ float red[256];
    const float* x = X + (long)blockIdx.x * xStride;
    float* y = Y + (long)blockIdx.x * yStride;
    const int tid = threadIdx.x;

    float s = 0.0f;
    for (int i = tid; i < n; i += 256) s += x[i];
    red[tid] = s; __syncthreads();
    for (int o = 128; o > 0; o >>= 1) { if (tid < o) red[tid] += red[tid + o]; __syncthreads(); }
    float mean = red[0] / n;
    __syncthreads();

    float v = 0.0f;
    for (int i = tid; i < n; i += 256) { float d = x[i] - mean; v += d * d; }
    red[tid] = v; __syncthreads();
    for (int o = 128; o > 0; o >>= 1) { if (tid < o) red[tid] += red[tid + o]; __syncthreads(); }
    float rstd = rsqrtf(red[0] / n + 1e-5f);

    for (int i = tid; i < n; i += 256)
        y[i] = (x[i] - mean) * rstd * g[i] + b[i];
}

// ---------------------------------------------------------------------------
// Softmax over rows of length n with row stride ld (in-place)
// ---------------------------------------------------------------------------
__global__ __launch_bounds__(128) void softmax_k(float* __restrict__ S, int n, int ld)
{
    __shared__ float red[128];
    float* s = S + (long)blockIdx.x * ld;
    const int tid = threadIdx.x;

    float mx = -1e30f;
    for (int i = tid; i < n; i += 128) mx = fmaxf(mx, s[i]);
    red[tid] = mx; __syncthreads();
    for (int o = 64; o > 0; o >>= 1) { if (tid < o) red[tid] = fmaxf(red[tid], red[tid + o]); __syncthreads(); }
    mx = red[0]; __syncthreads();

    float sum = 0.0f;
    for (int i = tid; i < n; i += 128) { float e = __expf(s[i] - mx); s[i] = e; sum += e; }
    red[tid] = sum; __syncthreads();
    for (int o = 64; o > 0; o >>= 1) { if (tid < o) red[tid] += red[tid + o]; __syncthreads(); }
    float inv = 1.0f / red[0];

    for (int i = tid; i < n; i += 128) s[i] *= inv;
}

// ---------------------------------------------------------------------------
// Patch extraction: img[16,3,224,224] -> patches[16*196][768] in (py,px,c) order
// ---------------------------------------------------------------------------
__global__ __launch_bounds__(256) void patch_extract_k(
    const float* __restrict__ img, float* __restrict__ P)
{
    const int bp = blockIdx.x;          // b*196 + p
    const int b = bp / 196, p = bp % 196;
    const int ph = p / 14, pw = p % 14;
    for (int e = threadIdx.x; e < 768; e += 256) {
        int c  = e % 3;
        int px = (e / 3) % 16;
        int py = e / 48;
        float v = img[(((long)b * 3 + c) * 224 + (ph * 16 + py)) * 224 + (pw * 16 + px)];
        P[(long)bp * 768 + e] = v;
    }
}

// ---------------------------------------------------------------------------
// Assemble tokens: x[b][0]=cls+pos[0]; x[b][1+p]=xp[b*196+p]+pos[1+p]
// ---------------------------------------------------------------------------
__global__ __launch_bounds__(256) void assemble_k(
    const float* __restrict__ xp, const float* __restrict__ cls,
    const float* __restrict__ pos, float* __restrict__ X)
{
    long idx = (long)blockIdx.x * 256 + threadIdx.x;
    const long total = (long)BATCH * TOK * DMODEL;
    if (idx >= total) return;
    int d = (int)(idx % DMODEL);
    long r = idx / DMODEL;
    int t = (int)(r % TOK);
    long b = r / TOK;
    float v = (t == 0) ? cls[d] : xp[((long)b * 196 + (t - 1)) * DMODEL + d];
    X[idx] = v + pos[(long)t * DMODEL + d];
}

// ---------------------------------------------------------------------------
// Host-side helpers
// ---------------------------------------------------------------------------
static inline void* wsCarve(char*& p, size_t bytes) {
    void* r = (void*)p;
    p += (bytes + 255) & ~(size_t)255;
    return r;
}

template<bool B_BF16, bool B_NK, bool DO_GELU>
static void launch_gemm(dim3 grid, hipStream_t s,
    const float* A, int lda, long sAb, long sAh,
    const void* B, int ldb, long sBb, long sBh,
    const float* bias, const float* resid, int ldr, long sRb,
    float* C, int ldc, long sCb, long sCh,
    int M, int N, int K, float alpha, int nh)
{
    gemm_wmma_bf16<B_BF16, B_NK, DO_GELU><<<grid, dim3(256), 0, s>>>(
        A, lda, sAb, sAh, B, ldb, sBb, sBh, bias, resid, ldr, sRb,
        C, ldc, sCb, sCh, M, N, K, alpha, nh);
}

extern "C" void kernel_launch(void* const* d_in, const int* in_sizes, int n_in,
                              void* d_out, int out_size, void* d_ws, size_t ws_size,
                              hipStream_t stream)
{
    const float* img     = (const float*)d_in[0];
    const float* p1g     = (const float*)d_in[1];
    const float* p1b     = (const float*)d_in[2];
    const float* patch_w = (const float*)d_in[3];
    const float* patch_b = (const float*)d_in[4];
    const float* p2g     = (const float*)d_in[5];
    const float* p2b     = (const float*)d_in[6];
    const float* pos     = (const float*)d_in[7];
    const float* clstok  = (const float*)d_in[8];
    const float* Wq      = (const float*)d_in[9];
    const float* Wk      = (const float*)d_in[10];
    const float* Wv      = (const float*)d_in[11];
    const float* Wo      = (const float*)d_in[12];
    const float* ffg     = (const float*)d_in[13];
    const float* ffb     = (const float*)d_in[14];
    const float* fw1     = (const float*)d_in[15];
    const float* fb1     = (const float*)d_in[16];
    const float* fw2     = (const float*)d_in[17];
    const float* fb2     = (const float*)d_in[18];
    const float* flg     = (const float*)d_in[19];
    const float* flb     = (const float*)d_in[20];
    const float* head_w  = (const float*)d_in[21];
    const float* head_b  = (const float*)d_in[22];
    float* out = (float*)d_out;

    const long R  = (long)BATCH * TOK;        // 3152 token rows
    const long DD = (long)DMODEL * DMODEL;
    const long DM = (long)DMODEL * MLPD;

    char* wp = (char*)d_ws;
    float* x1  = (float*)wsCarve(wp, R * DMODEL * 4);
    float* x2  = (float*)wsCarve(wp, R * DMODEL * 4);
    float* qb  = (float*)wsCarve(wp, R * DMODEL * 4);   // also patches / ctx
    float* kb  = (float*)wsCarve(wp, R * DMODEL * 4);   // also patch-GEMM out
    float* vb  = (float*)wsCarve(wp, R * DMODEL * 4);   // also LN(h) buffer
    float* hm  = (float*)wsCarve(wp, R * MLPD * 4);
    float* sc  = (float*)wsCarve(wp, (size_t)BATCH * NHEAD * TOK * SLD * 4);
    __bf16* wt = (__bf16*)wsCarve(wp, (size_t)DM * 2);
    float* clsb = (float*)wsCarve(wp, (size_t)BATCH * DMODEL * 4);

    // ---- patch embedding ----
    patch_extract_k<<<BATCH * 196, 256, 0, stream>>>(img, qb);
    layernorm_k<<<BATCH * 196, 256, 0, stream>>>(qb, DMODEL, qb, DMODEL, p1g, p1b, DMODEL);
    convert_transpose_k<<<(int)((DD + 255) / 256), 256, 0, stream>>>(patch_w, wt, DMODEL, DMODEL);
    launch_gemm<true, true, false>(dim3(6, 25, 1), stream,
        qb, DMODEL, 0, 0, wt, DMODEL, 0, 0, patch_b, nullptr, 0, 0,
        kb, DMODEL, 0, 0, BATCH * 196, DMODEL, DMODEL, 1.0f, 1);
    layernorm_k<<<BATCH * 196, 256, 0, stream>>>(kb, DMODEL, kb, DMODEL, p2g, p2b, DMODEL);
    assemble_k<<<(int)((R * DMODEL + 255) / 256), 256, 0, stream>>>(kb, clstok, pos, x1);

    const dim3 gRowsD(6, 25, 1);               // N=768, M=3152 (128x128 tiles)
    const long sTok = (long)TOK * DMODEL;      // per-batch activation stride
    const long sScB = (long)NHEAD * TOK * SLD; // per-batch score stride
    const long sScH = (long)TOK * SLD;         // per-head score stride

    for (int l = 0; l < 12; ++l) {
        // Q, K, V projections
        convert_transpose_k<<<(int)((DD + 255) / 256), 256, 0, stream>>>(Wq + l * DD, wt, DMODEL, DMODEL);
        launch_gemm<true, true, false>(gRowsD, stream,
            x1, DMODEL, 0, 0, wt, DMODEL, 0, 0, nullptr, nullptr, 0, 0,
            qb, DMODEL, 0, 0, (int)R, DMODEL, DMODEL, 1.0f, 1);
        convert_transpose_k<<<(int)((DD + 255) / 256), 256, 0, stream>>>(Wk + l * DD, wt, DMODEL, DMODEL);
        launch_gemm<true, true, false>(gRowsD, stream,
            x1, DMODEL, 0, 0, wt, DMODEL, 0, 0, nullptr, nullptr, 0, 0,
            kb, DMODEL, 0, 0, (int)R, DMODEL, DMODEL, 1.0f, 1);
        convert_transpose_k<<<(int)((DD + 255) / 256), 256, 0, stream>>>(Wv + l * DD, wt, DMODEL, DMODEL);
        launch_gemm<true, true, false>(gRowsD, stream,
            x1, DMODEL, 0, 0, wt, DMODEL, 0, 0, nullptr, nullptr, 0, 0,
            vb, DMODEL, 0, 0, (int)R, DMODEL, DMODEL, 1.0f, 1);

        // scores = Q K^T / 8  (K activation already [token][dim] == [N][K] layout)
        launch_gemm<false, true, false>(dim3(2, 2, BATCH * NHEAD), stream,
            qb, DMODEL, sTok, HDIM, kb, DMODEL, sTok, HDIM,
            nullptr, nullptr, 0, 0,
            sc, SLD, sScB, sScH, TOK, TOK, HDIM, 0.125f, NHEAD);
        softmax_k<<<BATCH * NHEAD * TOK, 128, 0, stream>>>(sc, TOK, SLD);

        // ctx = attn @ V   (V is [K][N] layout -> transpose-staging path), into qb
        launch_gemm<false, false, false>(dim3(1, 2, BATCH * NHEAD), stream,
            sc, SLD, sScB, sScH, vb, DMODEL, sTok, HDIM,
            nullptr, nullptr, 0, 0,
            qb, DMODEL, sTok, HDIM, TOK, HDIM, TOK, 1.0f, NHEAD);

        // x2 = ctx @ Wo + x1
        convert_transpose_k<<<(int)((DD + 255) / 256), 256, 0, stream>>>(Wo + l * DD, wt, DMODEL, DMODEL);
        launch_gemm<true, true, false>(gRowsD, stream,
            qb, DMODEL, 0, 0, wt, DMODEL, 0, 0, nullptr, x1, DMODEL, 0,
            x2, DMODEL, 0, 0, (int)R, DMODEL, DMODEL, 1.0f, 1);

        // MLP: h = gelu(LN(x2) @ W1 + b1); x1 = h @ W2 + b2 + x2
        layernorm_k<<<(int)R, 256, 0, stream>>>(x2, DMODEL, vb, DMODEL, ffg + l * DMODEL, ffb + l * DMODEL, DMODEL);
        convert_transpose_k<<<(int)((DM + 255) / 256), 256, 0, stream>>>(fw1 + l * DM, wt, DMODEL, MLPD);
        launch_gemm<true, true, true>(dim3(24, 25, 1), stream,
            vb, DMODEL, 0, 0, wt, DMODEL, 0, 0, fb1 + l * MLPD, nullptr, 0, 0,
            hm, MLPD, 0, 0, (int)R, MLPD, DMODEL, 1.0f, 1);
        convert_transpose_k<<<(int)((DM + 255) / 256), 256, 0, stream>>>(fw2 + l * DM, wt, MLPD, DMODEL);
        launch_gemm<true, true, false>(gRowsD, stream,
            hm, MLPD, 0, 0, wt, MLPD, 0, 0, fb2 + l * DMODEL, x2, DMODEL, 0,
            x1, DMODEL, 0, 0, (int)R, DMODEL, MLPD, 1.0f, 1);
    }

    // final LN on cls rows only, then head
    layernorm_k<<<BATCH, 256, 0, stream>>>(x1, (long)TOK * DMODEL, clsb, DMODEL, flg, flb, DMODEL);
    convert_transpose_k<<<(int)(((long)DMODEL * 1000 + 255) / 256), 256, 0, stream>>>(head_w, wt, DMODEL, 1000);
    launch_gemm<true, true, false>(dim3(8, 1, 1), stream,
        clsb, DMODEL, 0, 0, wt, DMODEL, 0, 0, head_b, nullptr, 0, 0,
        out, 1000, 0, 0, BATCH, 1000, DMODEL, 1.0f, 1);
}
